// SelfAttentionLayer_4612794876173
// MI455X (gfx1250) — matrix operand order repaired and checked by reading, not compiled, not run on hardware
//
#include <hip/hip_runtime.h>
#include <stdint.h>

#define BATCH  4
#define SEQ    2048
#define DMODEL 1024
#define NHEAD  16
#define HDIM   64
#define KT     64   // keys per attention iteration

typedef __bf16 bf16_t;
typedef __bf16 v16bf __attribute__((ext_vector_type(16)));
typedef __bf16 v8bf  __attribute__((ext_vector_type(8)));
typedef float  v8f   __attribute__((ext_vector_type(8)));
typedef unsigned int v4u __attribute__((ext_vector_type(4)));
typedef int v4i __attribute__((ext_vector_type(4)));
typedef int v8i __attribute__((ext_vector_type(8)));

static __device__ __forceinline__ v8f v8f_zero() {
  v8f z;
#pragma unroll
  for (int i = 0; i < 8; ++i) z[i] = 0.0f;
  return z;
}

static __device__ __forceinline__ v8f wmma_bf16(v16bf a, v16bf b, v8f c) {
  return __builtin_amdgcn_wmma_f32_16x16x32_bf16(false, a, false, b, (short)0, c,
                                                 false, false);
}

// Raw cross-lane xor-shuffle (no canonicalization): reads lane (lane^msk).
static __device__ __forceinline__ float xor_bperm(float v, int addr_bytes) {
  return __int_as_float(__builtin_amdgcn_ds_bpermute(addr_bytes, __float_as_int(v)));
}

// A fragment (16x32 bf16): lane L holds row (base_row + (L&15)).
// Lanes 0-15: K = {0..7, 16..23}; lanes 16-31: K = {8..15, 24..31}.
static __device__ __forceinline__ v16bf a_frag_bf16(const bf16_t* base, int ld,
                                                    int row, int k0, int lane) {
  const bf16_t* p = base + (size_t)row * ld + k0 + ((lane >> 4) << 3);
  union { v16bf v; v8bf h[2]; } u;
  u.h[0] = *(const v8bf*)(p);
  u.h[1] = *(const v8bf*)(p + 16);
  return u.v;
}

static __device__ __forceinline__ v16bf a_frag_f32(const float* __restrict__ base,
                                                   int ld, int row, int k0, int lane) {
  const float* p = base + (size_t)row * ld + k0 + ((lane >> 4) << 3);
  v16bf r;
#pragma unroll
  for (int i = 0; i < 8; ++i) r[i] = (__bf16)p[i];
#pragma unroll
  for (int i = 0; i < 8; ++i) r[8 + i] = (__bf16)p[16 + i];
  return r;
}

// B fragment (32x16 bf16): lane L holds column (base_col + (L&15)).
// Lanes 0-15: K = {0..7, 16..23}; lanes 16-31: K = {8..15, 24..31}.
static __device__ __forceinline__ v16bf b_frag_bf16(const bf16_t* base, int ld,
                                                    int col, int k0, int lane) {
  const bf16_t* p = base + (size_t)col * ld + k0 + ((lane >> 4) << 3);
  union { v16bf v; v8bf h[2]; } u;
  u.h[0] = *(const v8bf*)(p);
  u.h[1] = *(const v8bf*)(p + 16);
  return u.v;
}

// ---- Tensor Data Mover: 2D tile (bf16) global -> LDS, tracked by TENSORcnt.
// D# per CDNA5 ISA ch.8. 6-arg builtin: (v4u, v8i, v4i, v4i, v8i, i32 cpol).
static __device__ __forceinline__ void tdm_load_2d_bf16(
    unsigned lds_byte_addr, const void* gaddr, unsigned tile_d0, unsigned tile_d1,
    unsigned tensor_d0, unsigned tensor_d1, unsigned stride0_elems) {
  unsigned long long ga = (unsigned long long)(uintptr_t)gaddr;
  v4u g0;
  g0[0] = 1u;                                          // count=1, user descriptor
  g0[1] = lds_byte_addr;                               // lds_addr [63:32]
  g0[2] = (unsigned)(ga & 0xFFFFFFFFu);                // global_addr [95:64]
  g0[3] = (unsigned)((ga >> 32) & 0x01FFFFFFu) | (2u << 30);  // addr[56:32] | type=2
  v8i g1;
  g1[0] = (int)(1u << 16);                             // data_size=1 (2 bytes)
  g1[1] = (int)((tensor_d0 & 0xFFFFu) << 16);          // tensor_dim0[15:0] @ bits63:48
  g1[2] = (int)((tensor_d0 >> 16) & 0xFFFFu) |
          (int)((tensor_d1 & 0xFFFFu) << 16);          // dim0 hi | dim1 lo
  g1[3] = (int)((tensor_d1 >> 16) & 0xFFFFu) |
          (int)((tile_d0 & 0xFFFFu) << 16);            // dim1 hi | tile_dim0
  g1[4] = (int)(tile_d1 & 0xFFFFu);                    // tile_dim1 (tile_dim2=0)
  g1[5] = (int)stride0_elems;                          // tensor_dim0_stride lo32
  g1[6] = 0;
  g1[7] = 0;
  v4i z4 = {0, 0, 0, 0};
  v8i z8 = {0, 0, 0, 0, 0, 0, 0, 0};
  __builtin_amdgcn_tensor_load_to_lds(g0, g1, z4, z4, z8, 0);
}

static __device__ __forceinline__ unsigned lds_offset_of(const void* p) {
  return (unsigned)(uintptr_t)p;  // LDS aperture: low 32 bits are the LDS offset
}

__global__ void cvt_f32_bf16(const float* __restrict__ in, bf16_t* __restrict__ out,
                             int n) {
  int i = blockIdx.x * blockDim.x + threadIdx.x;
  if (i < n) out[i] = (__bf16)in[i];
}

// out[m,n] = (sum_k A[m,k]*W[n,k] + bias[n]) * scale
// MODE 0: A = f32, store bf16 in [B,H,S,HD]
// MODE 1: A = f32, store bf16 in [B,H,HD,S]   (transposed V for attention)
// MODE 2: A = bf16, store f32 row-major [B*S, D]  (final projection -> d_out)
template <int MODE>
__global__ __launch_bounds__(256) void proj_kernel(const void* __restrict__ Aptr,
                                                   const bf16_t* __restrict__ W,
                                                   const float* __restrict__ bias,
                                                   void* __restrict__ Out,
                                                   float scale) {
  const int lane = threadIdx.x & 31;
  const int wave = threadIdx.x >> 5;
  const int m0 = blockIdx.y * 256 + wave * 32;
  const int n0 = blockIdx.x * 64;
  const int l15 = lane & 15;

  v8f acc[2][4];
#pragma unroll
  for (int i = 0; i < 2; ++i)
#pragma unroll
    for (int j = 0; j < 4; ++j) acc[i][j] = v8f_zero();

  for (int k = 0; k < DMODEL; k += 32) {
    v16bf a0, a1;
    if (MODE == 2) {
      const bf16_t* A = (const bf16_t*)Aptr;
      a0 = a_frag_bf16(A, DMODEL, m0 + l15, k, lane);
      a1 = a_frag_bf16(A, DMODEL, m0 + 16 + l15, k, lane);
    } else {
      const float* A = (const float*)Aptr;
      a0 = a_frag_f32(A, DMODEL, m0 + l15, k, lane);
      a1 = a_frag_f32(A, DMODEL, m0 + 16 + l15, k, lane);
    }
    v16bf bfr[4];
#pragma unroll
    for (int j = 0; j < 4; ++j)
      bfr[j] = b_frag_bf16(W, DMODEL, n0 + j * 16 + l15, k, lane);
#pragma unroll
    for (int j = 0; j < 4; ++j) {
      acc[0][j] = wmma_bf16(a0, bfr[j], acc[0][j]);
      acc[1][j] = wmma_bf16(a1, bfr[j], acc[1][j]);
    }
  }

  const int rbase = (lane >> 4) << 3;
#pragma unroll
  for (int j = 0; j < 4; ++j) {
    const int n = n0 + j * 16 + l15;
    const float bn = bias[n];
#pragma unroll
    for (int i = 0; i < 2; ++i) {
#pragma unroll
      for (int r = 0; r < 8; ++r) {
        const int m = m0 + i * 16 + rbase + r;
        const float val = (acc[i][j][r] + bn) * scale;
        if (MODE == 2) {
          ((float*)Out)[(size_t)m * DMODEL + n] = val;
        } else {
          const int b = m >> 11, s = m & (SEQ - 1);
          const int h = n >> 6, d = n & (HDIM - 1);
          size_t addr;
          if (MODE == 0)
            addr = (((size_t)(b * NHEAD + h) * SEQ) + s) * HDIM + d;
          else
            addr = (((size_t)(b * NHEAD + h) * HDIM) + d) * SEQ + s;
          ((bf16_t*)Out)[addr] = (__bf16)val;
        }
      }
    }
  }
}

// Flash attention. Block = 8 waves sharing one (b,h), 128 query rows.
// 64-key tiles of K and V^T staged into double-buffered LDS by the Tensor
// Data Mover (wave 0 issues; TENSORcnt + barrier synchronize). The 1/sqrt(HD)
// scale is pre-folded into Q.
__global__ __launch_bounds__(256) void attn_kernel(const bf16_t* __restrict__ Qp,
                                                   const bf16_t* __restrict__ Kp,
                                                   const bf16_t* __restrict__ VT,
                                                   bf16_t* __restrict__ X) {
  __shared__ __align__(128) bf16_t kbuf[2][KT * HDIM];  // [keys][d]   8KB each
  __shared__ __align__(128) bf16_t vbuf[2][HDIM * KT];  // [d][keys]   8KB each
  __shared__ __align__(32)  bf16_t plds[8][16 * KT];    // per-wave P staging 2KB

  const int lane = threadIdx.x & 31;
  const int wave = threadIdx.x >> 5;
  const int l15 = lane & 15;
  const int bh = blockIdx.x >> 4;                 // 16 blocks per (b,h)
  const int q0 = ((blockIdx.x & 15) << 7) + (wave << 4);

  const bf16_t* Qb = Qp + (size_t)bh * SEQ * HDIM;
  const bf16_t* Kb = Kp + (size_t)bh * SEQ * HDIM;
  const bf16_t* Vb = VT + (size_t)bh * HDIM * SEQ;

  const v16bf aq0 = a_frag_bf16(Qb, HDIM, q0 + l15, 0, lane);
  const v16bf aq1 = a_frag_bf16(Qb, HDIM, q0 + l15, 32, lane);

  int xaddr[4];
#pragma unroll
  for (int i = 0; i < 4; ++i) xaddr[i] = (lane ^ (1 << i)) << 2;

  v8f acc[4];
  float mrow[8], lrow[8];
#pragma unroll
  for (int t = 0; t < 4; ++t) acc[t] = v8f_zero();
#pragma unroll
  for (int r = 0; r < 8; ++r) { mrow[r] = -1e30f; lrow[r] = 0.0f; }

  bf16_t* myp = &plds[wave][0];
  const int rbase = (lane >> 4) << 3;

  if (wave == 0) {  // prologue: DMA tile 0 into buffer 0
    tdm_load_2d_bf16(lds_offset_of(&kbuf[0][0]), Kb, HDIM, KT, HDIM, SEQ, HDIM);
    tdm_load_2d_bf16(lds_offset_of(&vbuf[0][0]), Vb, KT, HDIM, SEQ, HDIM, SEQ);
  }

  for (int it = 0; it < SEQ / KT; ++it) {
    const int cur = it & 1;
    __builtin_amdgcn_s_wait_tensorcnt(0);  // wave0 drains its TDM ops
    __syncthreads();                       // tile 'it' visible to all waves
    if (wave == 0 && it + 1 < SEQ / KT) {  // prefetch tile it+1 into other buf
      const int kn = (it + 1) * KT;
      tdm_load_2d_bf16(lds_offset_of(&kbuf[1 - cur][0]), Kb + (size_t)kn * HDIM,
                       HDIM, KT, HDIM, SEQ, HDIM);
      tdm_load_2d_bf16(lds_offset_of(&vbuf[1 - cur][0]), Vb + kn,
                       KT, HDIM, SEQ, HDIM, SEQ);
    }
    const bf16_t* kt_lds = &kbuf[cur][0];  // [64 keys][64 d]
    const bf16_t* vt_lds = &vbuf[cur][0];  // [64 d][64 keys]

    // S = Q K^T for 16 rows x 64 keys (four 16x16 C tiles), scale pre-folded
    v8f s[4];
#pragma unroll
    for (int j = 0; j < 4; ++j) {
      s[j] = v8f_zero();
      s[j] = wmma_bf16(aq0, b_frag_bf16(kt_lds, HDIM, j * 16 + l15, 0, lane), s[j]);
      s[j] = wmma_bf16(aq1, b_frag_bf16(kt_lds, HDIM, j * 16 + l15, 32, lane), s[j]);
    }

    // Online softmax: one max/sum reduction pass covers all 64 keys per row.
    float alpha[8];
#pragma unroll
    for (int r = 0; r < 8; ++r) {
      float mt = fmaxf(fmaxf(s[0][r], s[1][r]), fmaxf(s[2][r], s[3][r]));
#pragma unroll
      for (int i = 0; i < 4; ++i) mt = fmaxf(mt, xor_bperm(mt, xaddr[i]));
      const float mnew = fmaxf(mrow[r], mt);
      const float a = __expf(mrow[r] - mnew);
      mrow[r] = mnew;
      float ps = 0.0f;
#pragma unroll
      for (int j = 0; j < 4; ++j) {
        const float p = __expf(s[j][r] - mnew);
        s[j][r] = p;
        ps += p;
      }
#pragma unroll
      for (int i = 0; i < 4; ++i) ps += xor_bperm(ps, xaddr[i]);
      lrow[r] = lrow[r] * a + ps;
      alpha[r] = a;
    }
#pragma unroll
    for (int t = 0; t < 4; ++t)
#pragma unroll
      for (int r = 0; r < 8; ++r) acc[t][r] *= alpha[r];

    // C-layout -> A-fragment layout via per-wave LDS (DS in-order per wave).
#pragma unroll
    for (int j = 0; j < 4; ++j)
#pragma unroll
      for (int r = 0; r < 8; ++r)
        myp[(rbase + r) * KT + j * 16 + l15] = (__bf16)s[j][r];
    __builtin_amdgcn_wave_barrier();
    const v16bf pa0 = a_frag_bf16(myp, KT, l15, 0, lane);
    const v16bf pa1 = a_frag_bf16(myp, KT, l15, 32, lane);
    __builtin_amdgcn_wave_barrier();

    // acc += P (16x64) x V (64x64) from the LDS V^T tile
#pragma unroll
    for (int t = 0; t < 4; ++t) {
      acc[t] = wmma_bf16(pa0, b_frag_bf16(vt_lds, KT, t * 16 + l15, 0, lane), acc[t]);
      acc[t] = wmma_bf16(pa1, b_frag_bf16(vt_lds, KT, t * 16 + l15, 32, lane), acc[t]);
    }

    __syncthreads();  // everyone done reading buf[cur] before it is refilled
  }

  const int b = bh >> 4, h = bh & 15;
  float inv[8];
#pragma unroll
  for (int r = 0; r < 8; ++r) inv[r] = 1.0f / lrow[r];
#pragma unroll
  for (int t = 0; t < 4; ++t) {
#pragma unroll
    for (int r = 0; r < 8; ++r) {
      const int s = q0 + rbase + r;
      const int d = t * 16 + l15;
      X[((size_t)(b * SEQ + s)) * DMODEL + h * HDIM + d] =
          (__bf16)(acc[t][r] * inv[r]);
    }
  }
}

extern "C" void kernel_launch(void* const* d_in, const int* in_sizes, int n_in,
                              void* d_out, int out_size, void* d_ws, size_t ws_size,
                              hipStream_t stream) {
  (void)in_sizes; (void)n_in; (void)out_size; (void)ws_size;
  const float* q    = (const float*)d_in[0];
  const float* k    = (const float*)d_in[1];
  const float* v    = (const float*)d_in[2];
  const float* w_q  = (const float*)d_in[3];
  const float* b_q  = (const float*)d_in[4];
  const float* w_k  = (const float*)d_in[5];
  const float* b_k  = (const float*)d_in[6];
  const float* w_v  = (const float*)d_in[7];
  const float* b_v  = (const float*)d_in[8];
  const float* w_fc = (const float*)d_in[9];
  const float* b_fc = (const float*)d_in[10];

  char* ws = (char*)d_ws;
  const size_t WB = (size_t)DMODEL * DMODEL * sizeof(bf16_t);       // 2 MB
  const size_t PB = (size_t)BATCH * SEQ * DMODEL * sizeof(bf16_t);  // 16 MB
  bf16_t* wq_b  = (bf16_t*)(ws + 0 * WB);
  bf16_t* wk_b  = (bf16_t*)(ws + 1 * WB);
  bf16_t* wv_b  = (bf16_t*)(ws + 2 * WB);
  bf16_t* wfc_b = (bf16_t*)(ws + 3 * WB);
  bf16_t* Qp    = (bf16_t*)(ws + 4 * WB + 0 * PB);
  bf16_t* Kp    = (bf16_t*)(ws + 4 * WB + 1 * PB);
  bf16_t* VT    = (bf16_t*)(ws + 4 * WB + 2 * PB);
  bf16_t* X     = (bf16_t*)(ws + 4 * WB + 3 * PB);

  const int wn = DMODEL * DMODEL;
  cvt_f32_bf16<<<wn / 256, 256, 0, stream>>>(w_q, wq_b, wn);
  cvt_f32_bf16<<<wn / 256, 256, 0, stream>>>(w_k, wk_b, wn);
  cvt_f32_bf16<<<wn / 256, 256, 0, stream>>>(w_v, wv_b, wn);
  cvt_f32_bf16<<<wn / 256, 256, 0, stream>>>(w_fc, wfc_b, wn);

  dim3 pg(DMODEL / 64, (BATCH * SEQ) / 256);
  const float qscale = 0.125f;  // 1/sqrt(HD), folded into Q projection
  proj_kernel<0><<<pg, 256, 0, stream>>>(q, wq_b, b_q, Qp, qscale);
  proj_kernel<0><<<pg, 256, 0, stream>>>(k, wk_b, b_k, Kp, 1.0f);
  proj_kernel<1><<<pg, 256, 0, stream>>>(v, wv_b, b_v, VT, 1.0f);

  attn_kernel<<<(BATCH * NHEAD) * (SEQ / 128), 256, 0, stream>>>(Qp, Kp, VT, X);

  proj_kernel<2><<<pg, 256, 0, stream>>>(X, wfc_b, b_fc, d_out, 1.0f);
}
